// SwinAttention_72249939853980
// MI455X (gfx1250) — matrix operand order repaired and checked by reading, not compiled, not run on hardware
//
#include <hip/hip_runtime.h>
#include <hip/hip_bf16.h>

typedef _Float16 half_t;
typedef __attribute__((ext_vector_type(8)))  _Float16 v8h;
typedef __attribute__((ext_vector_type(16))) _Float16 v16h;
typedef __attribute__((ext_vector_type(8)))  float    v8f;

#define LDX 264   // halves per row of x/O tile   (64 rows)
#define LDQ 776   // halves per row of qkv tile   (64 rows, 768 cols)
#define LDP 72    // halves per row of P block    (16 rows per wave)
#define SMEM_BYTES (64*LDX*2 + 64*LDQ*2 + 8*16*LDP*2)

// ---------- WMMA fragment helpers (16x16x32 f16; A and B both use contiguous k = 16*(lane/16) + s) ----

// A-matrix 16x32 f16 from row-major [row][k] (ld halves/row, 16B-aligned rows).
// lanes 0-15: M=lane, K=0..15; lanes 16-31: M=lane-16, K=16..31
__device__ __forceinline__ v16h fragA(const half_t* base, int row0, int k0, int ld) {
    int lane = threadIdx.x & 31;
    const half_t* p = base + (row0 + (lane & 15)) * ld + k0 + ((lane & 16) ? 16 : 0);
    v8h lo = *(const v8h*)(p);        // K kb..kb+7
    v8h hi = *(const v8h*)(p + 8);    // K kb+8..kb+15
    return __builtin_shufflevector(lo, hi, 0,1,2,3,4,5,6,7,8,9,10,11,12,13,14,15);
}

// B-matrix 32x16 f16 from transposed source BT[n][k] (K contiguous per lane).
// lanes 0-15: N=lane, K=0..15; lanes 16-31: N=lane-16, K=16..31
__device__ __forceinline__ v16h fragB(const half_t* baseT, int col0, int k0, int ld) {
    int lane = threadIdx.x & 31;
    const half_t* p = baseT + (col0 + (lane & 15)) * ld + k0 + ((lane & 16) ? 16 : 0);
    v8h lo = *(const v8h*)(p);
    v8h hi = *(const v8h*)(p + 8);
    return __builtin_shufflevector(lo, hi, 0,1,2,3,4,5,6,7,8,9,10,11,12,13,14,15);
}

// B-matrix 32x16 f16 from row-major source S[k][n] (K is the strided dim).
__device__ __forceinline__ v16h fragB_strided(const half_t* base, int col0, int k0, int ld) {
    int lane = threadIdx.x & 31;
    int n  = col0 + (lane & 15);
    int kb = k0 + ((lane & 16) ? 16 : 0);
    v16h r;
#pragma unroll
    for (int i = 0; i < 16; ++i) r[i] = base[(kb + i) * ld + n];
    return r;
}

__device__ __forceinline__ v8f wmma16(v16h a, v16h b, v8f c) {
    return __builtin_amdgcn_wmma_f32_16x16x32_f16(false, a, false, b, (short)0, c, false, false);
}

// shift-mask region id for in-window flat index l, window coords (wi,wj)
__device__ __forceinline__ int regid(int l, int wi, int wj) {
    int r = l >> 3, c = l & 7;
    int rh = (wi == 15) ? ((r < 4) ? 1 : 2) : 0;
    int rw = (wj == 15) ? ((c < 4) ? 1 : 2) : 0;
    return rh * 3 + rw;
}

// ---------- prep kernels -----------------------------------------------------

// transpose + fp32->fp16 weights: qkv_wT[768][256], proj_wT[256][256]
__global__ void prep_weights(const float* __restrict__ qkv_w, const float* __restrict__ proj_w,
                             half_t* __restrict__ qkv_wT, half_t* __restrict__ proj_wT) {
    int idx = blockIdx.x * 256 + threadIdx.x;
    if (idx < 768 * 256) {
        int n = idx >> 8, k = idx & 255;
        qkv_wT[idx] = (half_t)qkv_w[k * 768 + n];
    } else {
        int j = idx - 768 * 256;
        int n = j >> 8, k = j & 255;
        proj_wT[j] = (half_t)proj_w[k * 256 + n];
    }
}

// Swin-v2 continuous relative position bias: rb[h][m][n], 16*sigmoid(MLP(log-coords))
__global__ void prep_bias(const float* __restrict__ w1, const float* __restrict__ b1,
                          const float* __restrict__ w2, float* __restrict__ rb) {
    int idx = blockIdx.x * 256 + threadIdx.x;   // 0..4095  (m*64+n)
    int m = idx >> 6, n = idx & 63;
    float d0 = (float)((m >> 3) - (n >> 3)) * (8.0f / 7.0f);
    float d1 = (float)((m & 7) - (n & 7)) * (8.0f / 7.0f);
    float rc0 = (d0 > 0.f ? 1.f : (d0 < 0.f ? -1.f : 0.f)) * __log2f(fabsf(d0) + 1.0f) * (1.0f / 3.0f);
    float rc1 = (d1 > 0.f ? 1.f : (d1 < 0.f ? -1.f : 0.f)) * __log2f(fabsf(d1) + 1.0f) * (1.0f / 3.0f);
    float acc[8];
#pragma unroll
    for (int h = 0; h < 8; ++h) acc[h] = 0.0f;
    for (int j = 0; j < 512; ++j) {
        float hj = fmaxf(rc0 * w1[j] + rc1 * w1[512 + j] + b1[j], 0.0f);
#pragma unroll
        for (int h = 0; h < 8; ++h) acc[h] += hj * w2[j * 8 + h];
    }
#pragma unroll
    for (int h = 0; h < 8; ++h)
        rb[h * 4096 + idx] = 16.0f / (1.0f + __expf(-acc[h]));
}

// ---------- fused window-attention kernel ------------------------------------
// 1 workgroup = 1 window (64 pixels). 8 waves; wave w owns head w.
__launch_bounds__(256, 1)
__global__ void swin_fused(const float* __restrict__ x,
                           const half_t* __restrict__ qkv_wT,
                           const float* __restrict__ q_bias,
                           const float* __restrict__ v_bias,
                           const float* __restrict__ scale,
                           const float* __restrict__ rel_bias,
                           const half_t* __restrict__ proj_wT,
                           const float* __restrict__ proj_b,
                           float* __restrict__ out) {
    extern __shared__ __align__(16) char smem[];
    half_t* lds_x = (half_t*)smem;                                // 64 x LDX (x tile, later O tile)
    half_t* lds_q = (half_t*)(smem + 64 * LDX * 2);               // 64 x LDQ (qkv)
    half_t* lds_p = (half_t*)(smem + 64 * LDX * 2 + 64 * LDQ * 2);// 8 waves x 16 x LDP (P block)

    const int tid  = threadIdx.x;
    const int lane = tid & 31;
    const int wv   = tid >> 5;          // wave id == head id
    const int nn0  = lane & 15;
    const int hi8  = (lane & 16) ? 8 : 0;
    const int bw   = blockIdx.x;
    const int b    = bw >> 8;
    const int win  = bw & 255;
    const int wi   = win >> 4, wj = win & 15;

    // ---- phase 0: gather rolled window of x -> f16 LDS ----
    {
        int l  = tid >> 2;
        int c0 = (tid & 3) * 64;
        int ph = ((wi << 3) + (l >> 3) + 4) & 127;
        int pw = ((wj << 3) + (l & 7) + 4) & 127;
        const float4* src = (const float4*)(x + (((size_t)b * 128 + ph) * 128 + pw) * 256 + c0);
        half_t* dst = lds_x + l * LDX + c0;
#pragma unroll
        for (int i = 0; i < 16; ++i) {
            float4 f = src[i];
            dst[4 * i + 0] = (half_t)f.x; dst[4 * i + 1] = (half_t)f.y;
            dst[4 * i + 2] = (half_t)f.z; dst[4 * i + 3] = (half_t)f.w;
        }
    }
    __syncthreads();

    // ---- phase 1: qkv GEMM  (wave w computes cols [96w, 96w+96) == head w's q,k,v) ----
#pragma unroll
    for (int rt = 0; rt < 4; ++rt) {
        v8f acc[6];
#pragma unroll
        for (int j = 0; j < 6; ++j) {
            int col = wv * 96 + j * 16 + nn0;
            float bv = 0.0f;
            if (col < 256) bv = q_bias[col];
            else if (col >= 512) bv = v_bias[col - 512];
#pragma unroll
            for (int i = 0; i < 8; ++i) acc[j][i] = bv;
        }
        for (int k = 0; k < 8; ++k) {
            v16h a = fragA(lds_x, rt * 16, k * 32, LDX);
#pragma unroll
            for (int j = 0; j < 6; ++j) {
                v16h bf = fragB(qkv_wT, wv * 96 + j * 16, k * 32, 256);
                acc[j] = wmma16(a, bf, acc[j]);
            }
        }
        int mrow = rt * 16 + hi8;
#pragma unroll
        for (int j = 0; j < 6; ++j) {
            int n = wv * 96 + j * 16 + nn0;
#pragma unroll
            for (int i = 0; i < 8; ++i)
                lds_q[(mrow + i) * LDQ + n] = (half_t)acc[j][i];
        }
    }
    __syncthreads();

    // ---- phase 2: l2-normalize q and k rows for this head (in place, f16) ----
    {
        half_t* qcol = lds_q + wv * 96;
        half_t* kcol = lds_q + wv * 96 + 32;
#pragma unroll
        for (int rr = 0; rr < 2; ++rr) {
            int l = lane + rr * 32;
            half_t* qp = qcol + l * LDQ;
            float s = 0.0f;
#pragma unroll
            for (int d = 0; d < 32; ++d) { float v = (float)qp[d]; s += v * v; }
            float r = rsqrtf(fmaxf(s, 1.55e-5f));
#pragma unroll
            for (int d = 0; d < 32; ++d) qp[d] = (half_t)((float)qp[d] * r);
            half_t* kp = kcol + l * LDQ;
            s = 0.0f;
#pragma unroll
            for (int d = 0; d < 32; ++d) { float v = (float)kp[d]; s += v * v; }
            r = rsqrtf(fmaxf(s, 1.55e-5f));
#pragma unroll
            for (int d = 0; d < 32; ++d) kp[d] = (half_t)((float)kp[d] * r);
        }
    }
    // no barrier: everything below uses only this wave's head slice + per-wave P

    // ---- phase 3: attention for head wv ----
    {
        const int h = wv;
        float ls = __expf(fminf(scale[h], 4.6051701859880914f)); // clamp at ln(100)
        const half_t* qbase = lds_q + h * 96;
        const half_t* kbase = lds_q + h * 96 + 32;
        const half_t* vbase = lds_q + h * 96 + 64;
        half_t* pbuf = lds_p + wv * 16 * LDP;

        for (int srt = 0; srt < 4; ++srt) {
            // S row-tile (16 x 64) = qn @ kn^T
            v8f sacc[4];
#pragma unroll
            for (int t = 0; t < 4; ++t)
#pragma unroll
                for (int i = 0; i < 8; ++i) sacc[t][i] = 0.0f;
            v16h aq = fragA(qbase, srt * 16, 0, LDQ);
#pragma unroll
            for (int t = 0; t < 4; ++t) {
                v16h bk = fragB(kbase, t * 16, 0, LDQ);   // k rows are BT[n][k]
                sacc[t] = wmma16(aq, bk, sacc[t]);
            }
            // scale + rel bias + shift mask, then softmax over 64 cols
#pragma unroll
            for (int i = 0; i < 8; ++i) {
                int mq   = srt * 16 + hi8 + i;
                int mreg = regid(mq, wi, wj);
                float mx = -1e30f;
#pragma unroll
                for (int t = 0; t < 4; ++t) {
                    int nq = t * 16 + nn0;
                    float s = ls * sacc[t][i]
                            + rel_bias[((h << 6) + mq) * 64 + nq]
                            + ((mreg != regid(nq, wi, wj)) ? -100.0f : 0.0f);
                    sacc[t][i] = s;
                    mx = fmaxf(mx, s);
                }
#pragma unroll
                for (int off = 1; off < 16; off <<= 1) mx = fmaxf(mx, __shfl_xor(mx, off, 32));
                float sum = 0.0f;
                float ex[4];
#pragma unroll
                for (int t = 0; t < 4; ++t) { ex[t] = __expf(sacc[t][i] - mx); sum += ex[t]; }
#pragma unroll
                for (int off = 1; off < 16; off <<= 1) sum += __shfl_xor(sum, off, 32);
                float inv = 1.0f / sum;
#pragma unroll
                for (int t = 0; t < 4; ++t)
                    pbuf[(hi8 + i) * LDP + t * 16 + nn0] = (half_t)(ex[t] * inv);
            }
            // O row-tile (16 x 32) = P @ v
            v8f oacc[2];
#pragma unroll
            for (int c = 0; c < 2; ++c)
#pragma unroll
                for (int i = 0; i < 8; ++i) oacc[c][i] = 0.0f;
#pragma unroll
            for (int kk = 0; kk < 2; ++kk) {
                v16h ap = fragA(pbuf, 0, kk * 32, LDP);
#pragma unroll
                for (int c = 0; c < 2; ++c) {
                    v16h bv = fragB_strided(vbase, c * 16, kk * 32, LDQ);
                    oacc[c] = wmma16(ap, bv, oacc[c]);
                }
            }
            int orow = srt * 16 + hi8;
#pragma unroll
            for (int c = 0; c < 2; ++c)
#pragma unroll
                for (int i = 0; i < 8; ++i)
                    lds_x[(orow + i) * LDX + h * 32 + c * 16 + nn0] = (half_t)oacc[c][i];
        }
    }
    __syncthreads();

    // ---- phase 4: output projection; store with reverse-shift gather ----
#pragma unroll
    for (int rt = 0; rt < 4; ++rt) {
        v8f pacc[2];
#pragma unroll
        for (int j = 0; j < 2; ++j) {
            float bv = proj_b[wv * 32 + j * 16 + nn0];
#pragma unroll
            for (int i = 0; i < 8; ++i) pacc[j][i] = bv;
        }
        for (int k = 0; k < 8; ++k) {
            v16h a = fragA(lds_x, rt * 16, k * 32, LDX);
#pragma unroll
            for (int j = 0; j < 2; ++j) {
                v16h bf = fragB(proj_wT, wv * 32 + j * 16, k * 32, 256);
                pacc[j] = wmma16(a, bf, pacc[j]);
            }
        }
        int mbase = rt * 16 + hi8;
#pragma unroll
        for (int i = 0; i < 8; ++i) {
            int l  = mbase + i;
            int ph = ((wi << 3) + (l >> 3) + 4) & 127;
            int pw = ((wj << 3) + (l & 7) + 4) & 127;
            float* op = out + (((size_t)b * 128 + ph) * 128 + pw) * 256 + wv * 32 + nn0;
            op[0]  = pacc[0][i];
            op[16] = pacc[1][i];
        }
    }
}

// ---------- host entry -------------------------------------------------------
extern "C" void kernel_launch(void* const* d_in, const int* in_sizes, int n_in,
                              void* d_out, int out_size, void* d_ws, size_t ws_size,
                              hipStream_t stream) {
    (void)in_sizes; (void)n_in; (void)out_size; (void)ws_size;
    const float* x      = (const float*)d_in[0];
    const float* qkv_w  = (const float*)d_in[1];
    const float* q_bias = (const float*)d_in[2];
    const float* v_bias = (const float*)d_in[3];
    const float* scale  = (const float*)d_in[4];
    const float* cpb_w1 = (const float*)d_in[5];
    const float* cpb_b1 = (const float*)d_in[6];
    const float* cpb_w2 = (const float*)d_in[7];
    const float* proj_w = (const float*)d_in[8];
    const float* proj_b = (const float*)d_in[9];
    float* out = (float*)d_out;

    char* ws = (char*)d_ws;
    float*  rb      = (float*)ws;                       // 8*64*64 f32   = 131072 B
    half_t* qkv_wT  = (half_t*)(ws + 131072);           // 768*256 f16   = 393216 B
    half_t* proj_wT = (half_t*)(ws + 131072 + 393216);  // 256*256 f16   = 131072 B

    prep_weights<<<1024, 256, 0, stream>>>(qkv_w, proj_w, qkv_wT, proj_wT);
    prep_bias<<<16, 256, 0, stream>>>(cpb_w1, cpb_b1, cpb_w2, rb);

    (void)hipFuncSetAttribute((const void*)swin_fused,
                              hipFuncAttributeMaxDynamicSharedMemorySize, SMEM_BYTES);
    swin_fused<<<4096, 256, SMEM_BYTES, stream>>>(x, qkv_wT, q_bias, v_bias, scale,
                                                  rb, proj_wT, proj_b, out);
}
